// STAG_Layer_3624952397870
// MI455X (gfx1250) — compile-verified, matrix operand order checked by reading, not run on hardware
//
#include <hip/hip_runtime.h>

// ---------------------------------------------------------------------------
// Types / fragment helpers
// ---------------------------------------------------------------------------
typedef __attribute__((ext_vector_type(16))) _Float16 v16h;
typedef __attribute__((ext_vector_type(8)))  float    v8f;

union Frag16 {
    v16h     v;
    _Float16 h[16];
    uint4    q[2];
};

// Load 16 f16 values for an A/B WMMA fragment: 8 contiguous halves at p,
// 8 contiguous halves at p+16 (the ISA's K {0..7} / {16..23} split, with the
// caller applying the lane>=16 offset of +8).
__device__ inline v16h frag_ld(const _Float16* p) {
    Frag16 f;
    f.q[0] = *reinterpret_cast<const uint4*>(p);
    f.q[1] = *reinterpret_cast<const uint4*>(p + 16);
    return f.v;
}

__device__ inline void atomAddF(float* p, float v) { unsafeAtomicAdd(p, v); }

// ---------------------------------------------------------------------------
// Weight transpose + f16 convert:  Wt[n*K + k] = (f16) W[k*Nc + n]
// ---------------------------------------------------------------------------
__global__ __launch_bounds__(256) void transpose_h_kernel(
    const float* __restrict__ W, _Float16* __restrict__ Wt, int K, int Nc) {
    int i = blockIdx.x * 256 + threadIdx.x;
    if (i >= K * Nc) return;
    int n = i / K, k = i - n * K;
    Wt[i] = (_Float16)W[k * Nc + n];
}

// ---------------------------------------------------------------------------
// h0 = concat(x, pseudo_coords)   (f32 residual copy + f16 GEMM copy)
// ---------------------------------------------------------------------------
__global__ __launch_bounds__(256) void concat_kernel(
    const float* __restrict__ x, const float* __restrict__ pc,
    float* __restrict__ h0f, _Float16* __restrict__ h0h, int N) {
    int i = blockIdx.x * 256 + threadIdx.x;
    if (i >= N * 128) return;
    int n = i >> 7, j = i & 127;
    float v = (j < 125) ? x[n * 125 + j] : pc[n * 3 + (j - 125)];
    h0f[i] = v;
    h0h[i] = (_Float16)v;
}

// ---------------------------------------------------------------------------
// Generic WMMA GEMM:  out[M,NOUT] = act(A[M,KDIM] @ W[KDIM,NOUT] + bias)
// W is staged as Wt[NOUT][KDIM] (f16, k-contiguous) in LDS.
// One wave computes a 16-row x NOUT tile; 8 waves/block -> 128 rows/block.
// ---------------------------------------------------------------------------
template <int KDIM, int NOUT, bool RELU>
__global__ __launch_bounds__(256) void gemm_kernel(
    const _Float16* __restrict__ A, const _Float16* __restrict__ Wt,
    const float* __restrict__ bias, const float* __restrict__ addin,
    float* __restrict__ outF, _Float16* __restrict__ outH, int M) {
    constexpr int PAD = (NOUT * KDIM <= 128 * 128) ? 8 : 0;  // LDS bank spread
    __shared__ _Float16 smW[NOUT * (KDIM + PAD)];
    for (int i = threadIdx.x; i < NOUT * KDIM; i += 256) {
        int n = i / KDIM, k = i - n * KDIM;
        smW[n * (KDIM + PAD) + k] = Wt[i];
    }
    __syncthreads();

    const int lane    = threadIdx.x & 31;
    const int wave    = threadIdx.x >> 5;
    const int rowBase = (blockIdx.x * 8 + wave) * 16;
    if (rowBase >= M) return;

    const int l15  = lane & 15;
    const int hi   = lane >> 4;   // 0: K {0..7,16..23}  1: K {8..15,24..31}
    const int koff = hi * 8;

    constexpr int NT = NOUT / 16;
    v8f acc[NT];
#pragma unroll
    for (int nt = 0; nt < NT; ++nt) acc[nt] = (v8f){0, 0, 0, 0, 0, 0, 0, 0};

    const _Float16* arow = A + (size_t)(rowBase + l15) * KDIM;
#pragma unroll
    for (int kt = 0; kt < KDIM / 32; ++kt) {
        v16h a = frag_ld(arow + kt * 32 + koff);
#pragma unroll
        for (int nt = 0; nt < NT; ++nt) {
            v16h b = frag_ld(&smW[(nt * 16 + l15) * (KDIM + PAD) + kt * 32 + koff]);
            acc[nt] = __builtin_amdgcn_wmma_f32_16x16x32_f16(
                false, a, false, b, (short)0, acc[nt], false, false);
        }
    }

#pragma unroll
    for (int nt = 0; nt < NT; ++nt) {
        int   col = nt * 16 + l15;
        float bb  = bias[col];
#pragma unroll
        for (int r = 0; r < 8; ++r) {
            int    m   = hi ? (8 + r) : r;
            size_t idx = (size_t)(rowBase + m) * NOUT + col;
            float  v   = acc[nt][r] + bb;
            if (RELU) v = v > 0.f ? v : 0.f;
            if (addin) v += addin[idx];
            if (outF) outF[idx] = v;
            if (outH) outH[idx] = (_Float16)v;
        }
    }
}

// ---------------------------------------------------------------------------
// Degree: deg[src[e]] += 1
// ---------------------------------------------------------------------------
__global__ __launch_bounds__(256) void deg_kernel(
    const int* __restrict__ srcI, float* __restrict__ deg, int E) {
    int e = blockIdx.x * 256 + threadIdx.x;
    if (e < E) atomAddF(&deg[srcI[e]], 1.0f);
}

// ---------------------------------------------------------------------------
// Fused edge kernel: per 16-edge tile (one wave):
//   t   = relu(rel_enc @ Wk1 + bk1)          (K=4, VALU, lane-local)
//   mod = t @ Wk2 + bk2                      (WMMA, Wk2^T from LDS)
//   sums[src] += h[dst] * mod                (fp32 atomics)
// ---------------------------------------------------------------------------
__global__ __launch_bounds__(256) void edge_kernel(
    const float* __restrict__ rel, const int* __restrict__ srcI,
    const int* __restrict__ dstI, const float* __restrict__ Wk1,
    const float* __restrict__ bk1, const _Float16* __restrict__ Wk2t,
    const float* __restrict__ bk2, const float* __restrict__ hnode,
    float* __restrict__ sums, int E) {
    constexpr int PAD = 8;
    __shared__ _Float16 smW[128 * (128 + PAD)];
    __shared__ float    smK1[4 * 128];
    __shared__ float    smB1[128];
    for (int i = threadIdx.x; i < 128 * 128; i += 256) {
        int n = i >> 7, k = i & 127;
        smW[n * (128 + PAD) + k] = Wk2t[i];
    }
    for (int i = threadIdx.x; i < 512; i += 256) smK1[i] = Wk1[i];
    if (threadIdx.x < 128) smB1[threadIdx.x] = bk1[threadIdx.x];
    __syncthreads();

    const int lane  = threadIdx.x & 31;
    const int wave  = threadIdx.x >> 5;
    const int ebase = (blockIdx.x * 8 + wave) * 16;
    if (ebase >= E) return;

    const int l15  = lane & 15;
    const int hi   = lane >> 4;
    const int koff = hi * 8;

    // stage-1 inputs for this lane's edge row
    const int e0 = ebase + l15;
    float r0 = rel[e0 * 4 + 0], r1 = rel[e0 * 4 + 1];
    float r2 = rel[e0 * 4 + 2], r3 = rel[e0 * 4 + 3];

    v8f acc[8];
#pragma unroll
    for (int nt = 0; nt < 8; ++nt) acc[nt] = (v8f){0, 0, 0, 0, 0, 0, 0, 0};

#pragma unroll
    for (int kt = 0; kt < 4; ++kt) {
        Frag16 a;
#pragma unroll
        for (int i = 0; i < 8; ++i) {
            int   c0 = kt * 32 + koff + i;
            float t0 = smB1[c0] + r0 * smK1[c0] + r1 * smK1[128 + c0] +
                       r2 * smK1[256 + c0] + r3 * smK1[384 + c0];
            a.h[i] = (_Float16)(t0 > 0.f ? t0 : 0.f);
            int   c1 = c0 + 16;
            float t1 = smB1[c1] + r0 * smK1[c1] + r1 * smK1[128 + c1] +
                       r2 * smK1[256 + c1] + r3 * smK1[384 + c1];
            a.h[8 + i] = (_Float16)(t1 > 0.f ? t1 : 0.f);
        }
#pragma unroll
        for (int nt = 0; nt < 8; ++nt) {
            v16h b = frag_ld(&smW[(nt * 16 + l15) * (128 + PAD) + kt * 32 + koff]);
            acc[nt] = __builtin_amdgcn_wmma_f32_16x16x32_f16(
                false, a.v, false, b, (short)0, acc[nt], false, false);
        }
    }

    // gather-modulate-scatter
#pragma unroll
    for (int nt = 0; nt < 8; ++nt) {
        int   col = nt * 16 + l15;
        float bb  = bk2[col];
#pragma unroll
        for (int r = 0; r < 8; ++r) {
            int   m  = hi ? (8 + r) : r;
            int   e  = ebase + m;
            int   d  = dstI[e];
            int   s  = srcI[e];
            float mv = acc[nt][r] + bb;
            float mf = hnode[(size_t)d * 128 + col] * mv;
            atomAddF(&sums[(size_t)s * 128 + col], mf);
        }
    }
}

// ---------------------------------------------------------------------------
// combine1: pre = sums/max(deg,1) * (c1 + sqrt(deg+1e-6)*c2) + h0
// plus per-channel sum/sumsq accumulation (for BN1).  In-place into `sums`.
// ---------------------------------------------------------------------------
__global__ __launch_bounds__(256) void combine1_kernel(
    float* __restrict__ sums, const float* __restrict__ deg,
    const float* __restrict__ c1, const float* __restrict__ c2,
    const float* __restrict__ h0f, float* __restrict__ stats, int N) {
    constexpr int ROWS = 256;
    const int col = threadIdx.x & 127;
    const int rh  = threadIdx.x >> 7;
    float cc1 = c1[col], cc2 = c2[col];
    float s = 0.f, sq = 0.f;
    int r0   = blockIdx.x * ROWS;
    int rend = (r0 + ROWS < N) ? (r0 + ROWS) : N;
    for (int r = r0 + rh; r < rend; r += 2) {
        float  dg    = deg[r];
        float  scale = (cc1 + sqrtf(dg + 1e-6f) * cc2) / fmaxf(dg, 1.0f);
        size_t idx   = (size_t)r * 128 + col;
        float  v     = sums[idx] * scale + h0f[idx];
        sums[idx]    = v;
        s += v;
        sq += v * v;
    }
    __shared__ float redS[256], redQ[256];
    redS[threadIdx.x] = s;
    redQ[threadIdx.x] = sq;
    __syncthreads();
    if (threadIdx.x < 128) {
        atomAddF(&stats[threadIdx.x], redS[threadIdx.x] + redS[threadIdx.x + 128]);
        atomAddF(&stats[128 + threadIdx.x], redQ[threadIdx.x] + redQ[threadIdx.x + 128]);
    }
}

// Per-channel sum/sumsq of an existing [N,128] buffer (for BN2).
__global__ __launch_bounds__(256) void stats_kernel(
    const float* __restrict__ buf, float* __restrict__ stats, int N) {
    constexpr int ROWS = 256;
    const int col = threadIdx.x & 127;
    const int rh  = threadIdx.x >> 7;
    float s = 0.f, sq = 0.f;
    int r0   = blockIdx.x * ROWS;
    int rend = (r0 + ROWS < N) ? (r0 + ROWS) : N;
    for (int r = r0 + rh; r < rend; r += 2) {
        float v = buf[(size_t)r * 128 + col];
        s += v;
        sq += v * v;
    }
    __shared__ float redS[256], redQ[256];
    redS[threadIdx.x] = s;
    redQ[threadIdx.x] = sq;
    __syncthreads();
    if (threadIdx.x < 128) {
        atomAddF(&stats[threadIdx.x], redS[threadIdx.x] + redS[threadIdx.x + 128]);
        atomAddF(&stats[128 + threadIdx.x], redQ[threadIdx.x] + redQ[threadIdx.x + 128]);
    }
}

__global__ void bn_finalize_kernel(const float* __restrict__ stats,
                                   float* __restrict__ mr, int N) {
    int j = threadIdx.x;
    if (j < 128) {
        float m = stats[j] / (float)N;
        float v = stats[128 + j] / (float)N - m * m;
        mr[j]       = m;
        mr[128 + j] = rsqrtf(v + 1e-5f);
    }
}

__global__ __launch_bounds__(256) void bn_apply_kernel(
    const float* __restrict__ pre, const float* __restrict__ mr,
    const float* __restrict__ g, const float* __restrict__ be,
    float* __restrict__ outF, _Float16* __restrict__ outH, int total) {
    int i = blockIdx.x * 256 + threadIdx.x;
    if (i >= total) return;
    int   col = i & 127;
    float v   = (pre[i] - mr[col]) * mr[128 + col] * g[col] + be[col];
    if (outF) outF[i] = v;
    if (outH) outH[i] = (_Float16)v;
}

// ---------------------------------------------------------------------------
// Host launcher
// ---------------------------------------------------------------------------
extern "C" void kernel_launch(void* const* d_in, const int* in_sizes, int n_in,
                              void* d_out, int out_size, void* d_ws,
                              size_t ws_size, hipStream_t stream) {
    const float* x    = (const float*)d_in[0];
    const float* rel  = (const float*)d_in[1];
    const float* pc   = (const float*)d_in[2];
    const float* Wd   = (const float*)d_in[3];
    const float* bd   = (const float*)d_in[4];
    const float* Wp   = (const float*)d_in[5];
    const float* bp   = (const float*)d_in[6];
    const float* Wk1  = (const float*)d_in[7];
    const float* bk1  = (const float*)d_in[8];
    const float* Wk2  = (const float*)d_in[9];
    const float* bk2  = (const float*)d_in[10];
    const float* c1   = (const float*)d_in[11];
    const float* c2   = (const float*)d_in[12];
    const float* Wf1  = (const float*)d_in[13];
    const float* bf1  = (const float*)d_in[14];
    const float* Wf2  = (const float*)d_in[15];
    const float* bf2  = (const float*)d_in[16];
    const float* g1   = (const float*)d_in[17];
    const float* be1  = (const float*)d_in[18];
    const float* g2   = (const float*)d_in[19];
    const float* be2  = (const float*)d_in[20];
    const int*   eidx = (const int*)d_in[21];

    const int N = in_sizes[0] / 125;
    const int E = in_sizes[1] / 4;
    const int* srcI = eidx;
    const int* dstI = eidx + E;

    // workspace carving (256B aligned)
    char* w = (char*)d_ws;
    auto  carve = [&](size_t bytes) {
        void* p = (void*)w;
        w += (bytes + 255) & ~(size_t)255;
        return p;
    };
    float*    h0f    = (float*)carve((size_t)N * 128 * 4);
    _Float16* h0h    = (_Float16*)carve((size_t)N * 128 * 2);
    _Float16* t1h    = (_Float16*)carve((size_t)N * 128 * 2);
    float*    hnode  = (float*)carve((size_t)N * 128 * 4);
    float*    sums   = (float*)carve((size_t)N * 128 * 4);  // later: pre1 in-place
    float*    deg    = (float*)carve((size_t)N * 4);
    float*    h1f    = (float*)carve((size_t)N * 128 * 4);
    _Float16* h1h    = (_Float16*)carve((size_t)N * 128 * 2);
    _Float16* uh     = (_Float16*)carve((size_t)N * 256 * 2);
    float*    stats1 = (float*)carve(256 * 4);
    float*    mr1    = (float*)carve(256 * 4);
    float*    stats2 = (float*)carve(256 * 4);
    float*    mr2    = (float*)carve(256 * 4);
    _Float16* WdT    = (_Float16*)carve(128 * 128 * 2);
    _Float16* WpT    = (_Float16*)carve(128 * 128 * 2);
    _Float16* Wk2T   = (_Float16*)carve(128 * 128 * 2);
    _Float16* Wf1T   = (_Float16*)carve(256 * 128 * 2);
    _Float16* Wf2T   = (_Float16*)carve(128 * 256 * 2);
    float*    outF   = (float*)d_out;

    hipMemsetAsync(sums, 0, (size_t)N * 128 * 4, stream);
    hipMemsetAsync(deg, 0, (size_t)N * 4, stream);
    hipMemsetAsync(stats1, 0, 256 * 4, stream);
    hipMemsetAsync(stats2, 0, 256 * 4, stream);

    // weights -> f16 transposed
    transpose_h_kernel<<<(128 * 128 + 255) / 256, 256, 0, stream>>>(Wd, WdT, 128, 128);
    transpose_h_kernel<<<(128 * 128 + 255) / 256, 256, 0, stream>>>(Wp, WpT, 128, 128);
    transpose_h_kernel<<<(128 * 128 + 255) / 256, 256, 0, stream>>>(Wk2, Wk2T, 128, 128);
    transpose_h_kernel<<<(256 * 128 + 255) / 256, 256, 0, stream>>>(Wf1, Wf1T, 128, 256);
    transpose_h_kernel<<<(256 * 128 + 255) / 256, 256, 0, stream>>>(Wf2, Wf2T, 256, 128);

    // h0 = concat(x, pc)
    concat_kernel<<<((size_t)N * 128 + 255) / 256, 256, 0, stream>>>(x, pc, h0f, h0h, N);

    const int gemmBlocks = (N + 127) / 128;
    // t1 = relu(h0 @ Wd + bd)
    gemm_kernel<128, 128, true><<<gemmBlocks, 256, 0, stream>>>(
        h0h, WdT, bd, nullptr, nullptr, t1h, N);
    // h = t1 @ Wp + bp
    gemm_kernel<128, 128, false><<<gemmBlocks, 256, 0, stream>>>(
        t1h, WpT, bp, nullptr, hnode, nullptr, N);

    // degree + fused edge path
    deg_kernel<<<(E + 255) / 256, 256, 0, stream>>>(srcI, deg, E);
    edge_kernel<<<(E + 127) / 128, 256, 0, stream>>>(
        rel, srcI, dstI, Wk1, bk1, Wk2T, bk2, hnode, sums, E);

    // scale + residual + BN1
    combine1_kernel<<<(N + 255) / 256, 256, 0, stream>>>(sums, deg, c1, c2, h0f, stats1, N);
    bn_finalize_kernel<<<1, 128, 0, stream>>>(stats1, mr1, N);
    bn_apply_kernel<<<((size_t)N * 128 + 255) / 256, 256, 0, stream>>>(
        sums, mr1, g1, be1, h1f, h1h, N * 128);

    // FFN
    gemm_kernel<128, 256, true><<<gemmBlocks, 256, 0, stream>>>(
        h1h, Wf1T, bf1, nullptr, nullptr, uh, N);
    gemm_kernel<256, 128, false><<<gemmBlocks, 256, 0, stream>>>(
        uh, Wf2T, bf2, /*addin=*/h1f, /*outF=*/outF, nullptr, N);

    // BN2 (in-place on d_out)
    stats_kernel<<<(N + 255) / 256, 256, 0, stream>>>(outF, stats2, N);
    bn_finalize_kernel<<<1, 128, 0, stream>>>(stats2, mr2, N);
    bn_apply_kernel<<<((size_t)N * 128 + 255) / 256, 256, 0, stream>>>(
        outF, mr2, g2, be2, outF, nullptr, N * 128);
}
